// ParallelExperts_40862318854390
// MI455X (gfx1250) — matrix operand order repaired and checked by reading, not compiled, not run on hardware
//
#include <hip/hip_runtime.h>
#include <stdint.h>

// ---------------------------------------------------------------------------
// MI455X (gfx1250) fused LoRA grouped GEMM (scattermoe).
// Compute-bound at ~68.7 GFLOP; BF16 WMMA (16x16x32) with f32 accumulate is
// ~8x the fp32 WMMA (16x16x4) rate, and halves L2-resident weight bytes.
// Weight tiles are staged via the Tensor Data Mover (one D# per tile),
// activation tiles via async global->LDS loads.
// ---------------------------------------------------------------------------

typedef unsigned short u16;
typedef __attribute__((ext_vector_type(8)))  u16    u16x8;
typedef __attribute__((ext_vector_type(16))) __bf16 v16bf;
typedef __attribute__((ext_vector_type(8)))  float  v8f;
typedef __attribute__((ext_vector_type(4)))  int    v4i;
typedef __attribute__((ext_vector_type(4)))  unsigned int u32x4;
typedef __attribute__((ext_vector_type(8)))  int    i32x8;
typedef __attribute__((ext_vector_type(4)))  int    i32x4;

#define E_     8
#define DIN_   2048
#define DOUT_  2048
#define NTOK_  4096
#define NK_    8192
#define R_     16
#define BM_    128
#define BN_    128
#define BK_    32
#define LDSK_  (BK_ + 8)       // padded row stride (80B, keeps 16B alignment)
#define MAXMB_ 72              // <= 8192/128 + 8 expert boundary blocks

#if defined(__AMDGCN__) && __has_builtin(__builtin_amdgcn_global_load_async_to_lds_b128)
#define ATHENA_ASYNC 1
#else
#define ATHENA_ASYNC 0
#endif

#if defined(__AMDGCN__) && __has_builtin(__builtin_amdgcn_tensor_load_to_lds)
#define ATHENA_TDM 1
#else
#define ATHENA_TDM 0
#endif

__device__ __forceinline__ u16 f2bf(float f) {
  __bf16 h = (__bf16)f;                     // RNE convert
  return __builtin_bit_cast(u16, h);
}

union FragBF { v16bf v; u16x8 h[2]; };

__device__ __forceinline__ void copy16(const u16* __restrict__ g, u16* l) {
#if ATHENA_ASYNC
  // param0 = int4 AS(1)* global source, param1 = int4 AS(3)* LDS dest.
  typedef __attribute__((address_space(1))) v4i* gp_t;
  typedef __attribute__((address_space(3))) v4i* lp_t;
  __builtin_amdgcn_global_load_async_to_lds_b128(
      (gp_t)(unsigned long long)(uintptr_t)g,
      (lp_t)(unsigned int)(uintptr_t)l, 0, 0);
#else
  *(u16x8*)l = *(const u16x8*)g;
#endif
}

__device__ __forceinline__ void wait_async_own() {
#if ATHENA_ASYNC
#if __has_builtin(__builtin_amdgcn_s_wait_asynccnt)
  __builtin_amdgcn_s_wait_asynccnt(0);
#else
  asm volatile("s_wait_asynccnt 0x0" ::: "memory");
#endif
#endif
}

// ---------------------------------------------------------------------------
// Pass 1: expert-aligned M-block table (one thread; 8 segments).
// ---------------------------------------------------------------------------
__global__ void map_kernel(const int* __restrict__ offs,
                           int* __restrict__ blk_rs, int* __restrict__ blk_re,
                           int* __restrict__ blk_e, int* __restrict__ nblk) {
  if (threadIdx.x == 0 && blockIdx.x == 0) {
    int cnt = 0, start = 0;
    for (int e = 0; e < E_; ++e) {
      int end = offs[e];
      for (int r = start; r < end; r += BM_) {
        blk_rs[cnt] = r; blk_re[cnt] = end; blk_e[cnt] = e; ++cnt;
      }
      start = end;
    }
    *nblk = cnt;
  }
}

// ---------------------------------------------------------------------------
// Pass 2: gather routed rows -> bf16, build inverse permutation.
// ---------------------------------------------------------------------------
__global__ __launch_bounds__(256) void gather_x_kernel(
    const float* __restrict__ inputs, const int* __restrict__ ssi,
    const int* __restrict__ kscalar, u16* __restrict__ Xb,
    int* __restrict__ inv) {
  const int row = blockIdx.x;                 // sorted slot
  const int kk  = kscalar[0];
  const int s   = ssi[row];
  if (threadIdx.x == 0) inv[s] = row;
  const int token = s / kk;
  const float* __restrict__ src = inputs + (size_t)token * DIN_;
  u16* __restrict__ dst = Xb + (size_t)row * DIN_;
  for (int c = threadIdx.x * 8; c < DIN_; c += 256 * 8) {
    float4 f0 = *(const float4*)(src + c);
    float4 f1 = *(const float4*)(src + c + 4);
    u16x8 u;
    u[0]=f2bf(f0.x); u[1]=f2bf(f0.y); u[2]=f2bf(f0.z); u[3]=f2bf(f0.w);
    u[4]=f2bf(f1.x); u[5]=f2bf(f1.y); u[6]=f2bf(f1.z); u[7]=f2bf(f1.w);
    *(u16x8*)(dst + c) = u;
  }
}

// ---------------------------------------------------------------------------
// Pass 3: fold LoRA into weights with WMMA:  W' = bf16(W + 2*B@A), K=16 padded
// to 32 with zeros. One 16x16 output tile per wave; 131072 tiles total.
// ---------------------------------------------------------------------------
__global__ __launch_bounds__(256) void fold_w_kernel(
    const float* __restrict__ weight, const float* __restrict__ lora_A,
    const float* __restrict__ lora_B, u16* __restrict__ Wp) {
  const int lane = threadIdx.x & 31;
  const int wave = threadIdx.x >> 5;
  const int id   = blockIdx.x * 8 + wave;     // exact grid: no bounds check
  const int e    = id >> 14;                  // 16384 tiles per expert
  const int rem  = id & 16383;
  const int tm   = rem >> 7;                  // dout tile (M)
  const int tn   = rem & 127;                 // din  tile (N)
  const int l15  = lane & 15;
  const bool lo  = lane < 16;

  const float* __restrict__ Bp = lora_B + (size_t)e * DOUT_ * R_;  // [dout][r]
  const float* __restrict__ Ap = lora_A + (size_t)e * R_ * DIN_;   // [r][din]

  FragBF a, b;
  // A-matrix frag (16-bit layout): lane row m=l15; element j -> K per ISA map.
#pragma unroll
  for (int j = 0; j < 16; ++j) {
    int K = j + ((j < 8) ? (lo ? 0 : 8) : (lo ? 8 : 16));
    float v = (K < R_) ? 2.0f * Bp[(size_t)(tm * 16 + l15) * R_ + K] : 0.0f;
    a.v[j] = (__bf16)v;
  }
  // B-matrix frag: lane col n=l15; element j -> K per ISA map (K>=16 padded 0).
#pragma unroll
  for (int j = 0; j < 16; ++j) {
    int r = j + ((j < 8) ? (lo ? 0 : 8) : (lo ? 8 : 16));
    float v = (r < R_) ? Ap[(size_t)r * DIN_ + tn * 16 + l15] : 0.0f;
    b.v[j] = (__bf16)v;
  }
  v8f c = {};
  c = __builtin_amdgcn_wmma_f32_16x16x32_bf16(false, a.v, false, b.v,
                                              (short)0, c, false, false);
  const float* __restrict__ Wsrc = weight + (size_t)e * DOUT_ * DIN_;
  u16* __restrict__ Wdst = Wp + (size_t)e * DOUT_ * DIN_;
#pragma unroll
  for (int j = 0; j < 8; ++j) {
    int m = tm * 16 + j + (lo ? 0 : 8);
    int n = tn * 16 + l15;
    float val = Wsrc[(size_t)m * DIN_ + n] + c[j];
    Wdst[(size_t)m * DIN_ + n] = f2bf(val);
  }
}

// ---------------------------------------------------------------------------
// Pass 4: grouped GEMM. Y[row, :] = Xb[row, :] @ Wp[e]^T for expert-aligned
// row blocks. 256 threads = 8 waves as 4(M)x2(N); each wave: 32x64 sub-tile.
// Double-buffered LDS; W tile staged by the Tensor Data Mover (one D# per
// tile, wave 0, TENSORcnt), X tile by async global->LDS loads (ASYNCcnt).
// ---------------------------------------------------------------------------
__global__ __launch_bounds__(256) void moe_gemm_kernel(
    const u16* __restrict__ Xb, const u16* __restrict__ Wp,
    float* __restrict__ Y, const int* __restrict__ blk_rs,
    const int* __restrict__ blk_re, const int* __restrict__ blk_e,
    const int* __restrict__ nblk) {
  if ((int)blockIdx.y >= *nblk) return;
  const int rs = blk_rs[blockIdx.y];
  const int re = blk_re[blockIdx.y];
  const int e  = blk_e[blockIdx.y];
  const int n0 = blockIdx.x * BN_;

  __shared__ __align__(16) u16 Xs[2][BM_][LDSK_];
  __shared__ __align__(16) u16 Ws[2][BN_][LDSK_];

  const int tid  = threadIdx.x;
  const int lane = tid & 31;
  const int wave = tid >> 5;
  const int mb   = (wave & 3) * 32;   // wave row base in tile
  const int nb   = (wave >> 2) * 64;  // wave col base in tile

  const u16* __restrict__ Wpe = Wp + (size_t)e * DOUT_ * DIN_;

  auto load_tiles = [&](int buf, int kk) {
    const int k0 = kk * BK_;
#pragma unroll
    for (int i = 0; i < 2; ++i) {
      int chunk = tid + i * 256;              // 512 16B-chunks per tile
      int r  = chunk >> 2;                    // tile row
      int kc = (chunk & 3) * 8;               // element offset in row
      int gr = rs + r; if (gr > NK_ - 1) gr = NK_ - 1;   // tail clamp
      copy16(Xb + (size_t)gr * DIN_ + k0 + kc, &Xs[buf][r][kc]);
#if !ATHENA_TDM
      copy16(Wpe + (size_t)(n0 + r) * DIN_ + k0 + kc, &Ws[buf][r][kc]);
#endif
    }
#if ATHENA_TDM
    if (wave == 0) {
      // Tensor DMA descriptor (ISA 8.3/8.4): 2D tile, 128 rows x 32 bf16,
      // row stride 2048 elems; LDS pad 4 DWORDs after every 16 DWORDs (=64B
      // row) to match the LDSK_ = BK_+8 padded layout.
      const unsigned int lds_base =
          (unsigned int)(uintptr_t)&Ws[buf][0][0];
      const unsigned long long ga =
          (unsigned long long)(uintptr_t)(Wpe + (size_t)n0 * DIN_ + k0);
      u32x4 g0;
      g0[0] = 1u;                                   // count=1, user mode
      g0[1] = lds_base;                             // lds_addr (bytes)
      g0[2] = (unsigned int)(ga & 0xFFFFFFFFu);     // global_addr[31:0]
      g0[3] = (unsigned int)((ga >> 32) & 0x01FFFFFFu) | (2u << 30); // type=2
      i32x8 g1;
      g1[0] = (1 << 16)      // data_size = 1 (2 bytes)
            | (1 << 20)      // pad_enable
            | (3 << 22)      // pad_interval: 16 DWORDs (one 64B tile row)
            | (3 << 25);     // pad_amount: 4 DWORDs (16B row padding)
      g1[1] = (int)((unsigned)DIN_ << 16);          // tensor_dim0 lo16
      g1[2] = (int)((unsigned)DOUT_ << 16);         // dim0 hi=0 | dim1 lo16
      g1[3] = (int)((unsigned)BK_ << 16);           // dim1 hi=0 | tile_dim0
      g1[4] = (int)(unsigned)BN_;                   // tile_dim1 | tile_dim2=0
      g1[5] = (int)(unsigned)DIN_;                  // tensor_dim0_stride lo32
      g1[6] = 0;                                    // stride hi | dim1_stride
      g1[7] = 0;
      i32x4 g2 = {0, 0, 0, 0};                      // 2D: groups 2/3 unused
      i32x4 g3 = {0, 0, 0, 0};
      i32x8 g4 = {0, 0, 0, 0, 0, 0, 0, 0};          // trailing group (unused)
      __builtin_amdgcn_tensor_load_to_lds(g0, g1, g2, g3, g4, 0);
    }
#endif
  };

  v8f zero = {};
  v8f acc[2][4];
#pragma unroll
  for (int tm = 0; tm < 2; ++tm)
#pragma unroll
    for (int tn = 0; tn < 4; ++tn) acc[tm][tn] = zero;

  load_tiles(0, 0);
  const int KSTEPS = DIN_ / BK_;              // 64
  const int l15 = lane & 15;
  const int ka  = (lane < 16) ? 0 : 8;        // A-frag K chunk base (ISA map)
  const int kb  = (lane >> 4) * 8;            // B-frag K chunk base (ISA map)

  for (int kk = 0; kk < KSTEPS; ++kk) {
    const int cur = kk & 1;
    wait_async_own();
#if ATHENA_TDM
    if (wave == 0) __builtin_amdgcn_s_wait_tensorcnt(0);
#endif
    __syncthreads();                          // buf `cur` ready everywhere
    if (kk + 1 < KSTEPS) load_tiles(cur ^ 1, kk + 1);

    FragBF a[2], b[4];
#pragma unroll
    for (int tm = 0; tm < 2; ++tm) {
      const u16* p = &Xs[cur][mb + tm * 16 + l15][0];
      a[tm].h[0] = *(const u16x8*)(p + ka);        // K = ka..ka+7
      a[tm].h[1] = *(const u16x8*)(p + ka + 16);   // K = ka+16..ka+23
    }
#pragma unroll
    for (int tn = 0; tn < 4; ++tn) {
      const u16* p = &Ws[cur][nb + tn * 16 + l15][0];
      b[tn].h[0] = *(const u16x8*)(p + kb);        // K = kb..kb+7
      b[tn].h[1] = *(const u16x8*)(p + kb + 16);   // K = kb+16..kb+23
    }
#pragma unroll
    for (int tm = 0; tm < 2; ++tm)
#pragma unroll
      for (int tn = 0; tn < 4; ++tn)
        acc[tm][tn] = __builtin_amdgcn_wmma_f32_16x16x32_bf16(
            false, a[tm].v, false, b[tn].v, (short)0, acc[tm][tn],
            false, false);
  }

  // Epilogue: write Y; skip ragged rows past segment end (owned by next block)
#pragma unroll
  for (int tm = 0; tm < 2; ++tm) {
#pragma unroll
    for (int j = 0; j < 8; ++j) {
      int grow = rs + mb + tm * 16 + j + ((lane < 16) ? 0 : 8);
      if (grow < re) {
#pragma unroll
        for (int tn = 0; tn < 4; ++tn) {
          Y[(size_t)grow * DOUT_ + n0 + nb + tn * 16 + l15] = acc[tm][tn][j];
        }
      }
    }
  }
}

// ---------------------------------------------------------------------------
// Pass 5: per-token gate-weighted combine (fixed slot order => deterministic).
// Fully overwrites d_out.
// ---------------------------------------------------------------------------
__global__ __launch_bounds__(256) void combine_kernel(
    const float* __restrict__ Y, const float* __restrict__ gates,
    const int* __restrict__ inv, const int* __restrict__ kscalar,
    float* __restrict__ out) {
  const int t  = blockIdx.x;
  const int kk = kscalar[0];
  for (int c = threadIdx.x * 4; c < DOUT_; c += 256 * 4) {
    float ox = 0.f, oy = 0.f, oz = 0.f, ow = 0.f;
    for (int s = 0; s < kk; ++s) {
      int   r = inv[t * kk + s];
      float g = gates[t * kk + s];
      float4 y = *(const float4*)(Y + (size_t)r * DOUT_ + c);
      ox += g * y.x; oy += g * y.y; oz += g * y.z; ow += g * y.w;
    }
    float4 o; o.x = ox; o.y = oy; o.z = oz; o.w = ow;
    *(float4*)(out + (size_t)t * DOUT_ + c) = o;
  }
}

// ---------------------------------------------------------------------------
extern "C" void kernel_launch(void* const* d_in, const int* in_sizes, int n_in,
                              void* d_out, int out_size, void* d_ws,
                              size_t ws_size, hipStream_t stream) {
  const float* inputs  = (const float*)d_in[0];
  const float* weight  = (const float*)d_in[1];
  const float* lora_A  = (const float*)d_in[2];
  const float* lora_B  = (const float*)d_in[3];
  const float* gates   = (const float*)d_in[4];
  const int*   ssi     = (const int*)d_in[6];
  const int*   offs    = (const int*)d_in[7];
  const int*   kscalar = (const int*)d_in[8];
  float*       out     = (float*)d_out;

  char* w = (char*)d_ws;
  u16* Wp = (u16*)w;  w += (size_t)E_ * DOUT_ * DIN_ * sizeof(u16);   // 64 MB
  u16* Xb = (u16*)w;  w += (size_t)NK_ * DIN_ * sizeof(u16);          // 32 MB
  float* Y = (float*)w; w += (size_t)NK_ * DOUT_ * sizeof(float);     // 64 MB
  int* inv    = (int*)w; w += NK_ * sizeof(int);
  int* blk_rs = (int*)w; w += MAXMB_ * sizeof(int);
  int* blk_re = (int*)w; w += MAXMB_ * sizeof(int);
  int* blk_e  = (int*)w; w += MAXMB_ * sizeof(int);
  int* nblk   = (int*)w; w += sizeof(int);

  map_kernel<<<1, 32, 0, stream>>>(offs, blk_rs, blk_re, blk_e, nblk);
  gather_x_kernel<<<NK_, 256, 0, stream>>>(inputs, ssi, kscalar, Xb, inv);
  fold_w_kernel<<<(E_ * 128 * 128) / 8, 256, 0, stream>>>(weight, lora_A,
                                                          lora_B, Wp);
  dim3 grid(DOUT_ / BN_, MAXMB_, 1);
  moe_gemm_kernel<<<grid, 256, 0, stream>>>(Xb, Wp, Y, blk_rs, blk_re, blk_e,
                                            nblk);
  combine_kernel<<<NTOK_, 256, 0, stream>>>(Y, gates, inv, kscalar, out);
}
